// AxialChannelAttention_33638183862617
// MI455X (gfx1250) — matrix operand order repaired and checked by reading, not compiled, not run on hardware
//
#include <hip/hip_runtime.h>
#include <hip/hip_bf16.h>
#include <math.h>

// Problem constants
#define B_  16
#define C_  256
#define CR_ 64
#define H_  128
#define W_  128
#define NEG_SLOPE 0.01f

typedef __attribute__((ext_vector_type(16))) _Float16 v16h;
typedef __attribute__((ext_vector_type(8)))  float    v8f;

// ---------------------------------------------------------------------------
// Kernel 1: axial pooling. One block per (b,c) plane (4096 blocks, 128 thr).
// Thread t owns column w=t: accumulates over H in registers (-> avg_h sums,
// max_h). Row reductions (over W per h) via wave32 shuffle + LDS combine.
// ---------------------------------------------------------------------------
__global__ __launch_bounds__(128) void ax_pool_kernel(
    const float* __restrict__ x,
    float* __restrict__ avgh_sum,   // [B*C, W]
    float* __restrict__ maxh,       // [B*C, W]
    float* __restrict__ avgw_sum,   // [B*C, H]
    float* __restrict__ maxw)       // [B*C, H]
{
    const int bc   = blockIdx.x;          // b*C + c
    const int t    = threadIdx.x;         // 0..127 == w
    const int wid  = t >> 5;
    const int lane = t & 31;

    const float* xp = x + (size_t)bc * (H_ * W_);

    __shared__ float rs[4][H_];
    __shared__ float rm[4][H_];

    float cs = 0.0f;
    float cm = -3.402823466e38f;

    for (int h = 0; h < H_; ++h) {
        float v = xp[h * W_ + t];
        cs += v;
        cm = fmaxf(cm, v);
        float s = v, m = v;
        #pragma unroll
        for (int off = 16; off > 0; off >>= 1) {
            s += __shfl_down(s, off, 32);
            m  = fmaxf(m, __shfl_down(m, off, 32));
        }
        if (lane == 0) { rs[wid][h] = s; rm[wid][h] = m; }
    }
    __syncthreads();

    avgh_sum[bc * W_ + t] = cs;
    maxh    [bc * W_ + t] = cm;
    avgw_sum[bc * H_ + t] = rs[0][t] + rs[1][t] + rs[2][t] + rs[3][t];
    maxw    [bc * H_ + t] = fmaxf(fmaxf(rm[0][t], rm[1][t]),
                                  fmaxf(rm[2][t], rm[3][t]));
}

// ---------------------------------------------------------------------------
// Kernel 2: first 1x1 conv applied to pooled tensors (separable through the
// broadcast-add). Pa[b,r,w] = (1/H)*sum_c w1[r,c]*avgh_sum[b,c,w], etc.
// 64 blocks = 4 arrays x 16 batches, 128 threads. Tiny, L2-resident.
// ---------------------------------------------------------------------------
__global__ __launch_bounds__(128) void ax_gemm1_kernel(
    const float* __restrict__ w1,
    const float* __restrict__ avgh_sum, const float* __restrict__ avgw_sum,
    const float* __restrict__ maxh,     const float* __restrict__ maxw,
    float* __restrict__ Pa, float* __restrict__ Qa,
    float* __restrict__ Pm, float* __restrict__ Qm)
{
    const int a = blockIdx.x >> 4;
    const int b = blockIdx.x & 15;
    const float* pool; float* dst; float scale;
    switch (a) {
        case 0:  pool = avgh_sum; dst = Pa; scale = 1.0f / (float)H_; break;
        case 1:  pool = avgw_sum; dst = Qa; scale = 1.0f / (float)W_; break;
        case 2:  pool = maxh;     dst = Pm; scale = 1.0f;             break;
        default: pool = maxw;     dst = Qm; scale = 1.0f;             break;
    }
    pool += (size_t)b * C_ * 128;
    dst  += (size_t)b * CR_ * 128;

    const int w = threadIdx.x;  // 0..127

    for (int rt = 0; rt < CR_ / 8; ++rt) {
        float acc[8] = {0.f,0.f,0.f,0.f,0.f,0.f,0.f,0.f};
        for (int c = 0; c < C_; ++c) {
            float p = pool[c * 128 + w];
            #pragma unroll
            for (int k = 0; k < 8; ++k)
                acc[k] += w1[(rt * 8 + k) * C_ + c] * p;
        }
        #pragma unroll
        for (int k = 0; k < 8; ++k)
            dst[(rt * 8 + k) * 128 + w] = acc[k] * scale;
    }
}

// ---------------------------------------------------------------------------
// Kernel 3: pack w2 (C x CR fp32) into f16 in the CDNA5 WMMA 16-bit A-matrix
// 16x32 lane layout (CDNA5 ISA 7.12.2): lane L: m=L%16, half=L/16;
// v16h element e -> K = (e/8)*16 + half*8 + ((e/2)&3)*2 + (e&1).
// One contiguous 32B load per lane per (ctile,khalf).
// ---------------------------------------------------------------------------
__global__ __launch_bounds__(256) void ax_packw2_kernel(
    const float* __restrict__ w2, _Float16* __restrict__ w2p)
{
    const int idx = blockIdx.x * 256 + threadIdx.x;   // 0..16383
    const int e    = idx & 15;
    const int L    = (idx >> 4) & 31;
    const int kh   = (idx >> 9) & 1;
    const int ct   = idx >> 10;                       // channel tile 0..15
    const int half = L >> 4;
    const int m    = L & 15;
    const int K    = (e >> 3) * 16 + half * 8 + ((e >> 1) & 3) * 2 + (e & 1);
    const int r    = kh * 32 + K;
    const int c    = ct * 16 + m;
    w2p[idx] = (_Float16)w2[c * CR_ + r];
}

// ---------------------------------------------------------------------------
// Kernel 4 (hot): fused second GEMM (WMMA) + sigmoid + x*(1+s).
// One block (256 thr = 8 waves) per 32-pixel group: fixed (b,h), 32 consec w
// split into two 16-pixel WMMA N-groups that share the A-tiles.
//  - h_sum stored in LDS ALREADY in f16 WMMA-B lane layout:
//      value (r,n) -> hsB[pg][r>=32][((r%32)/8%2)*16 + n][((r%32)/16)*8 + r%8]
//    row padded to 24 halves (48B: 16B-aligned for ds_load_b128, 2-way banks)
//  - consumer B fetch: one aligned 32B vector load per (pg,khalf)
//  - each wave: 2 M-iter x 2 pixel-groups x 2 WMMAs (K=64) = 8 v_wmma
// ---------------------------------------------------------------------------
__global__ __launch_bounds__(256) void ax_fused_kernel(
    const float* __restrict__ x,
    const float* __restrict__ Pa, const float* __restrict__ Qa,
    const float* __restrict__ Pm, const float* __restrict__ Qm,
    const _Float16* __restrict__ w2p,
    float* __restrict__ out)
{
    __shared__ __align__(16) _Float16 hsB[2][2][32][24];  // pg, khalf, lane, e

    const int blk = blockIdx.x;                 // B * H * (W/32) = 8192
    const int b   = blk >> 9;                   // H*(W/32) = 512
    const int rem = blk & 511;
    const int h   = rem >> 2;                   // W/32 = 4
    const int w0  = (rem & 3) << 5;             // 0,32,64,96

    const int tid = threadIdx.x;

    const float* Pab = Pa + (size_t)b * CR_ * 128;
    const float* Qab = Qa + (size_t)b * CR_ * 128;
    const float* Pmb = Pm + (size_t)b * CR_ * 128;
    const float* Qmb = Qm + (size_t)b * CR_ * 128;

    // ---- build h_sum: 2048 values (32 pixels x 64 r), 8 per thread,
    //      converted to f16 and scattered into WMMA-B layout ----
    #pragma unroll
    for (int i = 0; i < 8; ++i) {
        int idx = tid + i * 256;        // 0..2047
        int pg  = idx >> 10;            // pixel group 0/1
        int r   = (idx >> 4) & 63;
        int nn  = idx & 15;
        int w   = w0 + (pg << 4) + nn;
        float va = Pab[r * 128 + w] + Qab[r * 128 + h];
        float vm = Pmb[r * 128 + w] + Qmb[r * 128 + h];
        va = (va >= 0.f) ? va : NEG_SLOPE * va;   // leaky relu
        vm = (vm >= 0.f) ? vm : NEG_SLOPE * vm;
        int rr = r & 31;
        hsB[pg][r >> 5][(((rr >> 3) & 1) << 4) + nn][((rr >> 4) << 3) + (rr & 7)] = (_Float16)(va + vm);
    }
    __syncthreads();

    const int lane = tid & 31;
    const int wid  = tid >> 5;
    const int n    = lane & 15;
    const int half = lane >> 4;

    // ---- B operands: aligned 32B vector loads straight from LDS ----
    v16h bm[2][2];
    #pragma unroll
    for (int pg = 0; pg < 2; ++pg) {
        bm[pg][0] = *(const v16h*)&hsB[pg][0][lane][0];   // K  0..31
        bm[pg][1] = *(const v16h*)&hsB[pg][1][lane][0];   // K 32..63
    }

    const v16h* Ap = (const v16h*)w2p;

    #pragma unroll
    for (int mi = 0; mi < 2; ++mi) {
        const int ct = mi * 8 + wid;                // channel tile 0..15
        v16h a0 = Ap[(ct * 2 + 0) * 32 + lane];     // K  0..31 of w2 tile
        v16h a1 = Ap[(ct * 2 + 1) * 32 + lane];     // K 32..63 of w2 tile
        const int c0 = ct << 4;

        #pragma unroll
        for (int pg = 0; pg < 2; ++pg) {
            v8f acc = {};
            acc = __builtin_amdgcn_wmma_f32_16x16x32_f16(
                      false, a0, false, bm[pg][0], (short)0, acc, false, false);
            acc = __builtin_amdgcn_wmma_f32_16x16x32_f16(
                      false, a1, false, bm[pg][1], (short)0, acc, false, false);

            // ---- epilogue: sigmoid gate + residual scaling, fp32 ----
            const int wcol = w0 + (pg << 4) + n;
            #pragma unroll
            for (int i = 0; i < 8; ++i) {
                int c = c0 + i + (half << 3);       // C/D layout: M = i + 8*half
                size_t off = ((((size_t)b * C_ + c) * H_) + h) * W_ + wcol;
                float xv = x[off];
                float g  = acc[i];
                float sg = 1.0f / (1.0f + __expf(-g));
                out[off] = xv * (1.0f + sg);
            }
        }
    }
}

// ---------------------------------------------------------------------------
// Launch
// ---------------------------------------------------------------------------
extern "C" void kernel_launch(void* const* d_in, const int* in_sizes, int n_in,
                              void* d_out, int out_size, void* d_ws, size_t ws_size,
                              hipStream_t stream) {
    const float* x  = (const float*)d_in[0];   // [B,C,H,W]
    const float* w1 = (const float*)d_in[1];   // [CR,C]
    const float* w2 = (const float*)d_in[2];   // [C,CR]
    float* out = (float*)d_out;

    // workspace carve-up (all offsets 32B-aligned)
    float* avgh = (float*)d_ws;                         // B*C*W
    float* maxh = avgh + (size_t)B_ * C_ * W_;
    float* avgw = maxh + (size_t)B_ * C_ * W_;
    float* maxw = avgw + (size_t)B_ * C_ * H_;
    float* Pa   = maxw + (size_t)B_ * C_ * H_;          // B*CR*W
    float* Qa   = Pa + (size_t)B_ * CR_ * W_;
    float* Pm   = Qa + (size_t)B_ * CR_ * H_;
    float* Qm   = Pm + (size_t)B_ * CR_ * W_;
    _Float16* w2p = (_Float16*)(Qm + (size_t)B_ * CR_ * H_);  // 16384 halves

    ax_pool_kernel<<<B_ * C_, 128, 0, stream>>>(x, avgh, maxh, avgw, maxw);
    ax_gemm1_kernel<<<4 * B_, 128, 0, stream>>>(w1, avgh, avgw, maxh, maxw,
                                                Pa, Qa, Pm, Qm);
    ax_packw2_kernel<<<64, 256, 0, stream>>>(w2, w2p);
    ax_fused_kernel<<<B_ * H_ * (W_ / 32), 256, 0, stream>>>(
        x, Pa, Qa, Pm, Qm, w2p, out);
}
